// Seq2SeqAttn2DecoderUF_90348932038993
// MI455X (gfx1250) — matrix operand order repaired and checked by reading, not compiled, not run on hardware
//
#include <hip/hip_runtime.h>
#include <hip/hip_bf16.h>
#include <math.h>

// Problem constants (from reference)
#define B_   64
#define SE_  512
#define HD_  1024
#define ED_  512
#define N_   10000
#define T_   20
#define M_   1536
#define G4_  4096
#define EOS_ 1
#define NEG_BIG_  (-1000000000.0f)
#define MASK_VAL_ (-10000000.0f)

typedef __attribute__((ext_vector_type(16))) _Float16 v16h;
typedef __attribute__((ext_vector_type(8)))  _Float16 v8h;
typedef __attribute__((ext_vector_type(8)))  float    v8f;

static __device__ __forceinline__ float sigmoidf_(float x) {
    return 1.0f / (1.0f + __expf(-x));
}
static __device__ __forceinline__ float waveSum(float v) {
    #pragma unroll
    for (int m = 16; m > 0; m >>= 1) v += __shfl_xor(v, m, 32);
    return v;
}

// ---------------- conversion / init ----------------
__global__ void k_f32_to_f16(const float* __restrict__ src, _Float16* __restrict__ dst, long n) {
    long i = (long)blockIdx.x * blockDim.x + threadIdx.x;
    long stride = (long)gridDim.x * blockDim.x;
    for (; i < n; i += stride) dst[i] = (_Float16)src[i];
}
__global__ void k_zero_f32(float* p, long n) {
    long i = (long)blockIdx.x * blockDim.x + threadIdx.x;
    long stride = (long)gridDim.x * blockDim.x;
    for (; i < n; i += stride) p[i] = 0.0f;
}
__global__ void k_zero_f16(_Float16* p, long n) {
    long i = (long)blockIdx.x * blockDim.x + threadIdx.x;
    long stride = (long)gridDim.x * blockDim.x;
    for (; i < n; i += stride) p[i] = (_Float16)0.0f;
}
__global__ void k_init_state(const float* __restrict__ hidden, const float* __restrict__ cell,
                             float* __restrict__ h, float* __restrict__ c,
                             _Float16* __restrict__ h_h) {
    int i = blockIdx.x * blockDim.x + threadIdx.x;
    if (i >= B_ * HD_) return;
    float hv = hidden[i];
    h[i] = hv;
    c[i] = cell[i];
    h_h[i] = (_Float16)hv;
}

// ---------------- generic WMMA GEMM (64-row M-strip per wave) ----------------
// out[m][n] = act( (acc? out[m][n] : 0) + sum_k A[m][k]*W[n][k] + bias[n] + mask[m][n] )
// A: Mrows x K (f16, row-major, lda=K); W: Nout x K (f16, row-major, ldw=K)
// Each wave: one 16-col N tile x four 16-row M tiles (B loaded once, 4 WMMAs per k-step).
__global__ __launch_bounds__(128)
void k_gemm_wmma(const _Float16* __restrict__ A,
                 const _Float16* __restrict__ W,
                 const float* __restrict__ bias,       // may be null
                 const float* __restrict__ addmask,    // may be null
                 int ldmask,
                 float* __restrict__ out,              // may be null
                 long ldo,
                 _Float16* __restrict__ out_h,         // may be null
                 int ldoh,
                 int K, int Nout,
                 int accumulate, int relu) {
    const int ntiles = (Nout + 15) >> 4;
    const int wave = threadIdx.x >> 5;
    const int nTile = blockIdx.x * 4 + wave;
    if (nTile >= ntiles) return;               // uniform per wave
    const int m0 = blockIdx.y * 64;            // 64-row M strip

    const int lane = threadIdx.x & 31;
    const int l16  = lane & 15;
    const bool hi  = lane >= 16;

    int ncol = nTile * 16 + l16;
    const int ncl = (ncol < Nout) ? ncol : (Nout - 1);   // clamp for ragged tiles

    const _Float16* aBase0 = A + (size_t)(m0 + l16) * K + (hi ? 8 : 0);
    const _Float16* bBase  = W + (size_t)ncl * K + (hi ? 8 : 0);

    v8f acc0 = {}, acc1 = {}, acc2 = {}, acc3 = {};
    for (int k0 = 0; k0 < K; k0 += 32) {
        // B tile: two 8-half runs per lane (k = 8h.., 16+8h..), loaded once per k-step
        v8h blo = *(const v8h*)(bBase + k0);
        v8h bhi = *(const v8h*)(bBase + k0 + 16);
        v16h b;
        #pragma unroll
        for (int i = 0; i < 8; ++i) { b[i] = blo[i]; b[i + 8] = bhi[i]; }
        if (k0 + 128 < K) __builtin_prefetch(bBase + k0 + 128, 0, 1);

        // four A tiles of the 64-row strip reuse this B tile
        #pragma unroll
        for (int mt = 0; mt < 4; ++mt) {
            const _Float16* aB = aBase0 + (size_t)(mt * 16) * K + k0;
            v8h alo = *(const v8h*)(aB);
            v8h ahi = *(const v8h*)(aB + 16);
            v16h a;
            #pragma unroll
            for (int i = 0; i < 8; ++i) { a[i] = alo[i]; a[i + 8] = ahi[i]; }
            v8f* accp = mt == 0 ? &acc0 : mt == 1 ? &acc1 : mt == 2 ? &acc2 : &acc3;
            *accp = __builtin_amdgcn_wmma_f32_16x16x32_f16(
                false, a, false, b, (short)0, *accp, false, false);
        }
    }

    const float bv = (bias && ncol < Nout) ? bias[ncol] : 0.0f;
    #pragma unroll
    for (int mt = 0; mt < 4; ++mt) {
        v8f acc = mt == 0 ? acc0 : mt == 1 ? acc1 : mt == 2 ? acc2 : acc3;
        #pragma unroll
        for (int r = 0; r < 8; ++r) {
            const int m = m0 + mt * 16 + r + (hi ? 8 : 0);
            if (ncol >= Nout) continue;
            float v = acc[r] + bv;
            if (addmask)    v += addmask[(size_t)m * ldmask + ncol];
            if (accumulate && out) v += out[(size_t)m * ldo + ncol];
            if (relu)       v = v > 0.0f ? v : 0.0f;
            if (out)        out[(size_t)m * ldo + ncol] = v;
            if (out_h)      out_h[(size_t)m * ldoh + ncol] = (_Float16)v;
        }
    }
}

// ---------------- LSTM pointwise ----------------
__global__ void k_lstm(const float* __restrict__ gates,
                       float* __restrict__ h, float* __restrict__ c,
                       _Float16* __restrict__ h_h) {
    int i = blockIdx.x * blockDim.x + threadIdx.x;
    if (i >= B_ * HD_) return;
    int b = i / HD_, j = i - b * HD_;
    const float* g = gates + (size_t)b * G4_;
    float ig = g[j], fg = g[HD_ + j], gg = g[2 * HD_ + j], og = g[3 * HD_ + j];
    float cn = sigmoidf_(fg) * c[i] + sigmoidf_(ig) * tanhf(gg);
    float hn = sigmoidf_(og) * tanhf(cn);
    c[i] = cn; h[i] = hn; h_h[i] = (_Float16)hn;
}

// ---------------- encoder attention ----------------
// one wave per (b, s): score[b][s] = dot(enc[b][s], q[b]) + pad
__global__ void k_score(const _Float16* __restrict__ enc_h,
                        const float* __restrict__ q,
                        const int* __restrict__ text_lengths,
                        float* __restrict__ score) {
    int row = blockIdx.x * (blockDim.x >> 5) + (threadIdx.x >> 5);
    if (row >= B_ * SE_) return;
    int b = row / SE_, s = row - b * SE_;
    int lane = threadIdx.x & 31;
    const _Float16* e = enc_h + (size_t)row * HD_;
    const float* qb = q + (size_t)b * HD_;
    float acc = 0.0f;
    #pragma unroll
    for (int it = 0; it < 4; ++it) {
        int off = lane * 8 + it * 256;
        v8h ev = *(const v8h*)(e + off);
        #pragma unroll
        for (int j = 0; j < 8; ++j) acc += (float)ev[j] * qb[off + j];
    }
    acc = waveSum(acc);
    if (lane == 0) {
        float pad = (s < text_lengths[b]) ? 0.0f : NEG_BIG_;
        score[row] = acc + pad;
    }
}

// softmax over SE per batch row (in place)
__global__ void k_softmax_w(float* __restrict__ score) {
    const int b = blockIdx.x, tid = threadIdx.x;
    __shared__ float red[256];
    float* row = score + (size_t)b * SE_;
    float v0 = row[tid], v1 = row[tid + 256];
    float mx = fmaxf(v0, v1);
    red[tid] = mx; __syncthreads();
    for (int s = 128; s > 0; s >>= 1) {
        if (tid < s) red[tid] = fmaxf(red[tid], red[tid + s]);
        __syncthreads();
    }
    mx = red[0]; __syncthreads();
    float e0 = __expf(v0 - mx), e1 = __expf(v1 - mx);
    red[tid] = e0 + e1; __syncthreads();
    for (int s = 128; s > 0; s >>= 1) {
        if (tid < s) red[tid] += red[tid + s];
        __syncthreads();
    }
    float inv = 1.0f / red[0];
    row[tid] = e0 * inv; row[tid + 256] = e1 * inv;
}

// attn_enc[b][h] = sum_s w[b][s]*enc[b][s][h]
__global__ void k_attnenc(const float* __restrict__ w,
                          const _Float16* __restrict__ enc_h,
                          float* __restrict__ attn_enc) {
    const int b = blockIdx.x, tid = threadIdx.x;
    __shared__ float ws[256];
    float acc[4] = {0, 0, 0, 0};
    for (int s0 = 0; s0 < SE_; s0 += 256) {
        ws[tid] = w[(size_t)b * SE_ + s0 + tid];
        __syncthreads();
        for (int s = 0; s < 256; ++s) {
            float wv = ws[s];
            const _Float16* e = enc_h + ((size_t)(b * SE_ + s0 + s)) * HD_;
            #pragma unroll
            for (int j = 0; j < 4; ++j) acc[j] += wv * (float)e[tid + j * 256];
        }
        __syncthreads();
    }
    #pragma unroll
    for (int j = 0; j < 4; ++j) attn_enc[(size_t)b * HD_ + tid + j * 256] = acc[j];
}

// ---------------- buffer attention (last-token attention) ----------------
// k[t] = dot(kh[b][t], w_ko)+b_ko (masked), softmax over T, attn_last = sum_t sw[t]*buf[b][t]
__global__ void k_ksc_attnlast(const float* __restrict__ kh,
                               const float* __restrict__ w_ko,
                               const float* __restrict__ b_ko,
                               const _Float16* __restrict__ buf_h,
                               float* __restrict__ attn_last,
                               int step) {
    const int b = blockIdx.x, tid = threadIdx.x;
    const int wave = tid >> 5, lane = tid & 31;
    __shared__ float kls[T_];
    __shared__ float swl[T_];
    const int maxt = step > 1 ? step : 1;
    for (int t = wave; t < T_; t += 8) {
        const float* row = kh + ((size_t)(b * T_ + t)) * ED_;
        float acc = 0.0f;
        for (int j = lane; j < ED_; j += 32) acc += row[j] * w_ko[j];
        acc = waveSum(acc);
        if (lane == 0) {
            float kv = acc + b_ko[0];
            kls[t] = (t < maxt) ? kv : NEG_BIG_;
        }
    }
    __syncthreads();
    if (tid == 0) {
        float mx = kls[0];
        for (int t = 1; t < T_; ++t) mx = fmaxf(mx, kls[t]);
        float sum = 0.0f;
        for (int t = 0; t < T_; ++t) { swl[t] = __expf(kls[t] - mx); sum += swl[t]; }
        float inv = 1.0f / sum;
        for (int t = 0; t < T_; ++t) swl[t] *= inv;
    }
    __syncthreads();
    for (int e = tid; e < ED_; e += 256) {
        float acc = 0.0f;
        #pragma unroll
        for (int t = 0; t < T_; ++t)
            acc += swl[t] * (float)buf_h[((size_t)(b * T_ + t)) * ED_ + e];
        attn_last[(size_t)b * ED_ + e] = acc;
    }
}

// m = concat(sent + attn_enc, inp + attn_last) -> f16
__global__ void k_build_m(const float* __restrict__ enc,
                          const float* __restrict__ attn_enc,
                          const float* __restrict__ inp,
                          const float* __restrict__ attn_last,
                          _Float16* __restrict__ m_h) {
    int i = blockIdx.x * blockDim.x + threadIdx.x;
    if (i >= B_ * M_) return;
    int b = i / M_, j = i - b * M_;
    float v;
    if (j < HD_) v = enc[((size_t)b * SE_) * HD_ + j] + attn_enc[(size_t)b * HD_ + j];
    else         v = inp[(size_t)b * ED_ + (j - HD_)] + attn_last[(size_t)b * ED_ + (j - HD_)];
    m_h[i] = (_Float16)v;
}

// ---------------- argmax + state update ----------------
__global__ void k_argmax_update(const float* __restrict__ logits,  // d_out + t*N, ld = T*N
                                const float* __restrict__ embedding,
                                float* __restrict__ logits_mask,
                                float* __restrict__ inp,
                                _Float16* __restrict__ inp_h,
                                _Float16* __restrict__ buf_h,
                                float* __restrict__ ids_out,        // d_out + B*T*N
                                int t) {
    const int b = blockIdx.x, tid = threadIdx.x;
    __shared__ float bv[256];
    __shared__ int   bi[256];
    __shared__ int   sid;
    const float* row = logits + (size_t)b * (T_ * (long)N_);
    float best = -INFINITY; int bidx = N_;
    for (int n = tid; n < N_; n += 256) {
        float v = row[n];
        if (v > best || (v == best && n < bidx)) { best = v; bidx = n; }
    }
    bv[tid] = best; bi[tid] = bidx; __syncthreads();
    for (int s = 128; s > 0; s >>= 1) {
        if (tid < s) {
            if (bv[tid + s] > bv[tid] ||
                (bv[tid + s] == bv[tid] && bi[tid + s] < bi[tid])) {
                bv[tid] = bv[tid + s]; bi[tid] = bi[tid + s];
            }
        }
        __syncthreads();
    }
    if (tid == 0) {
        sid = bi[0];
        logits_mask[(size_t)b * N_ + sid]  = MASK_VAL_;
        logits_mask[(size_t)b * N_ + EOS_] = 0.0f;
        ids_out[(size_t)b * T_ + t] = (float)sid;
    }
    __syncthreads();
    const int id = sid;
    for (int e = tid; e < ED_; e += 256) {
        float ev = embedding[(size_t)id * ED_ + e];
        inp[(size_t)b * ED_ + e]   = ev;
        inp_h[(size_t)b * ED_ + e] = (_Float16)ev;
        buf_h[((size_t)(b * T_ + t)) * ED_ + e] = (_Float16)ev;
    }
}

// ---------------- host side ----------------
static inline size_t align256(size_t x) { return (x + 255) & ~(size_t)255; }

extern "C" void kernel_launch(void* const* d_in, const int* in_sizes, int n_in,
                              void* d_out, int out_size, void* d_ws, size_t ws_size,
                              hipStream_t stream) {
    const float* enc    = (const float*)d_in[0];
    const float* hidden = (const float*)d_in[1];
    const float* cell   = (const float*)d_in[2];
    const int*   tlen   = (const int*)  d_in[3];
    // d_in[4] encoder_mask (unused; derived from text_lengths)
    const float* embedding = (const float*)d_in[5];
    const float* w_ih = (const float*)d_in[6];
    const float* w_hh = (const float*)d_in[7];
    const float* b_ih = (const float*)d_in[8];
    const float* b_hh = (const float*)d_in[9];
    const float* w_q  = (const float*)d_in[10];
    const float* b_q  = (const float*)d_in[11];
    const float* w_k  = (const float*)d_in[12];
    const float* b_k  = (const float*)d_in[13];
    const float* w_ko = (const float*)d_in[14];
    const float* b_ko = (const float*)d_in[15];
    const float* w_m1 = (const float*)d_in[16];
    const float* b_m1 = (const float*)d_in[17];
    const float* w_m2 = (const float*)d_in[18];
    const float* b_m2 = (const float*)d_in[19];

    float* out = (float*)d_out;
    float* ids_out = out + (size_t)B_ * T_ * N_;

    // workspace carve-up
    char* ws = (char*)d_ws;
    size_t off = 0;
    auto take = [&](size_t bytes) { char* p = ws + off; off = align256(off + bytes); return p; };
    _Float16* enc_h  = (_Float16*)take((size_t)B_ * SE_ * HD_ * 2);
    _Float16* wih_h  = (_Float16*)take((size_t)G4_ * ED_ * 2);
    _Float16* whh_h  = (_Float16*)take((size_t)G4_ * HD_ * 2);
    _Float16* wq_h   = (_Float16*)take((size_t)HD_ * HD_ * 2);
    _Float16* wk_h   = (_Float16*)take((size_t)ED_ * ED_ * 2);
    _Float16* wm1_h  = (_Float16*)take((size_t)M_ * M_ * 2);
    _Float16* wm2_h  = (_Float16*)take((size_t)N_ * M_ * 2);
    float*    h      = (float*)take((size_t)B_ * HD_ * 4);
    float*    c      = (float*)take((size_t)B_ * HD_ * 4);
    _Float16* h_h    = (_Float16*)take((size_t)B_ * HD_ * 2);
    float*    inp    = (float*)take((size_t)B_ * ED_ * 4);
    _Float16* inp_h  = (_Float16*)take((size_t)B_ * ED_ * 2);
    float*    gates  = (float*)take((size_t)B_ * G4_ * 4);
    float*    q      = (float*)take((size_t)B_ * HD_ * 4);
    float*    score  = (float*)take((size_t)B_ * SE_ * 4);
    float*    attn_enc  = (float*)take((size_t)B_ * HD_ * 4);
    _Float16* buf_h  = (_Float16*)take((size_t)B_ * T_ * ED_ * 2);
    float*    kh     = (float*)take((size_t)B_ * T_ * ED_ * 4);
    float*    attn_last = (float*)take((size_t)B_ * ED_ * 4);
    _Float16* m_h    = (_Float16*)take((size_t)B_ * M_ * 2);
    _Float16* hid1_h = (_Float16*)take((size_t)B_ * M_ * 2);
    float*    logits_mask = (float*)take((size_t)B_ * N_ * 4);
    (void)ws_size; (void)in_sizes; (void)n_in; (void)out_size;

    auto cvt = [&](const float* s, _Float16* d, long n) {
        int grid = (int)((n + 255) / 256); if (grid > 65535) grid = 65535;
        k_f32_to_f16<<<grid, 256, 0, stream>>>(s, d, n);
    };
    // one-time (per launch) f16 weight staging
    cvt(enc,  enc_h,  (long)B_ * SE_ * HD_);
    cvt(w_ih, wih_h, (long)G4_ * ED_);
    cvt(w_hh, whh_h, (long)G4_ * HD_);
    cvt(w_q,  wq_h,  (long)HD_ * HD_);
    cvt(w_k,  wk_h,  (long)ED_ * ED_);
    cvt(w_m1, wm1_h, (long)M_ * M_);
    cvt(w_m2, wm2_h, (long)N_ * M_);

    // state init (must happen every call)
    k_init_state<<<(B_ * HD_ + 255) / 256, 256, 0, stream>>>(hidden, cell, h, c, h_h);
    k_zero_f32<<<256, 256, 0, stream>>>(logits_mask, (long)B_ * N_);
    k_zero_f32<<<64, 256, 0, stream>>>(inp, (long)B_ * ED_);
    k_zero_f16<<<64, 256, 0, stream>>>(inp_h, (long)B_ * ED_);
    k_zero_f16<<<256, 256, 0, stream>>>(buf_h, (long)B_ * T_ * ED_);

    const dim3 gblk(128);
    auto gemm = [&](const _Float16* A, const _Float16* W, const float* bias,
                    const float* mask, int ldmask,
                    float* o, long ldo, _Float16* oh, int ldoh,
                    int K, int Nout, int Mrows, int accum, int relu) {
        dim3 grid(((Nout + 15) / 16 + 3) / 4, Mrows / 64);
        k_gemm_wmma<<<grid, gblk, 0, stream>>>(A, W, bias, mask, ldmask,
                                               o, ldo, oh, ldoh, K, Nout, accum, relu);
    };

    for (int t = 0; t < T_; ++t) {
        // LSTM gates: inp @ w_ih^T + b_ih, then += h @ w_hh^T + b_hh
        gemm(inp_h, wih_h, b_ih, nullptr, 0, gates, G4_, nullptr, 0, ED_, G4_, B_, 0, 0);
        gemm(h_h,   whh_h, b_hh, nullptr, 0, gates, G4_, nullptr, 0, HD_, G4_, B_, 1, 0);
        k_lstm<<<(B_ * HD_ + 255) / 256, 256, 0, stream>>>(gates, h, c, h_h);

        // q = h @ w_q^T + b_q
        gemm(h_h, wq_h, b_q, nullptr, 0, q, HD_, nullptr, 0, HD_, HD_, B_, 0, 0);

        // encoder attention
        k_score<<<(B_ * SE_) / 8, 256, 0, stream>>>(enc_h, q, tlen, score);
        k_softmax_w<<<B_, 256, 0, stream>>>(score);
        k_attnenc<<<B_, 256, 0, stream>>>(score, enc_h, attn_enc);

        // buffer attention: kh = relu(buf @ w_k^T + b_k)  (rows = B*T)
        gemm(buf_h, wk_h, b_k, nullptr, 0, kh, ED_, nullptr, 0, ED_, ED_, B_ * T_, 0, 1);
        k_ksc_attnlast<<<B_, 256, 0, stream>>>(kh, w_ko, b_ko, buf_h, attn_last, t);

        // m, MLP, logits (mask folded into logits GEMM epilogue, written to d_out)
        k_build_m<<<(B_ * M_ + 255) / 256, 256, 0, stream>>>(enc, attn_enc, inp, attn_last, m_h);
        gemm(m_h, wm1_h, b_m1, nullptr, 0, nullptr, 0, hid1_h, M_, M_, M_, B_, 0, 1);
        gemm(hid1_h, wm2_h, b_m2, logits_mask, N_,
             out + (size_t)t * N_, (long)T_ * N_, nullptr, 0, M_, N_, B_, 0, 0);

        // argmax -> next ids, embedding feedback, mask/buffer update
        k_argmax_update<<<B_, 256, 0, stream>>>(out + (size_t)t * N_, embedding,
                                                logits_mask, inp, inp_h, buf_h, ids_out, t);
    }
}
